// LaneEncoder_3212635537618
// MI455X (gfx1250) — compile-verified, matrix operand order checked by reading, not run on hardware
//
#include <hip/hip_runtime.h>
#include <hip/hip_bf16.h>
#include <math.h>

// ---------------------------------------------------------------------------
// Problem constants (from reference)
// ---------------------------------------------------------------------------
#define BB 32
#define MM 256
#define LL 20
#define DD 256
#define HH 8
#define DHH 32
#define FF_ 1024
#define NN (BB * MM)        // 8192 lanes
#define NL (NN * LL)        // 163840 rows
#define NEGBIAS -1000000000.0f

typedef __attribute__((ext_vector_type(16))) _Float16 v16h;
typedef __attribute__((ext_vector_type(8)))  _Float16 v8h;
typedef __attribute__((ext_vector_type(8)))  float    v8f;

// ---------------------------------------------------------------------------
// helpers
// ---------------------------------------------------------------------------
__device__ inline float block_sum256(float v, float* red) {
    int tid = threadIdx.x;
    red[tid] = v;
    __syncthreads();
#pragma unroll
    for (int s = 128; s > 0; s >>= 1) {
        if (tid < s) red[tid] += red[tid + s];
        __syncthreads();
    }
    float r = red[0];
    __syncthreads();
    return r;
}

// Load a 16x32 f16 fragment row-slice for this lane from row-major K-contig
// memory.  p already points at row*K + k0 + (hiHalf ? 8 : 0).
// Low lanes get K {0..7,16..23}, high lanes K {8..15,24..31} per ISA layout.
__device__ inline v16h frag_load(const _Float16* __restrict__ p) {
    v8h lo = *reinterpret_cast<const v8h*>(p);        // 16B load
    v8h hi = *reinterpret_cast<const v8h*>(p + 16);   // 16B load
    v16h r;
#pragma unroll
    for (int i = 0; i < 8; ++i) { r[i] = lo[i]; r[i + 8] = hi[i]; }
    return r;
}

__device__ inline v8f wmma_f16(v16h a, v16h b, v8f c) {
    return __builtin_amdgcn_wmma_f32_16x16x32_f16(
        false, a, false, b, (short)0, c, false, false);
}

// ---------------------------------------------------------------------------
// WMMA GEMM:  C[rows x NOUT] = A[rows x K] * W[NOUT x K]^T
// K and NOUT are compile-time so every fragment load is base + immediate
// offset (good s_clause groups, no per-iteration 64-bit address adds).
// Each wave owns a 32x64 tile: 2 row-tiles x 4 col-tiles (8 accumulators),
// 12 b128 loads per K-step for 8 WMMAs.  Block = 8 waves = 64 rows x 256 cols.
// Epilogue flags: bit0 = +bias, bit1 = GeLU(erf), bit2 = +resid (fp32).
// ---------------------------------------------------------------------------
template <int K, int NOUT>
__global__ void gemm16_wmma(const _Float16* __restrict__ A,
                            const _Float16* __restrict__ W,
                            const float* __restrict__ bias,
                            const float* __restrict__ resid,
                            float* __restrict__ out32,
                            _Float16* __restrict__ out16,
                            int flags) {
    const unsigned tid  = threadIdx.x;
    const unsigned wave = tid >> 5;            // 0..7
    const unsigned lane = tid & 31u;
    const unsigned m    = lane & 15u;
    const bool hiHalf   = lane >= 16;
    const int rowPair   = (int)(wave >> 2);    // 0..1
    const int colQuad   = (int)(wave & 3u) + blockIdx.x * 4;
    const int col0      = colQuad * 64;
    if (col0 >= NOUT) return;
    const int row0 = blockIdx.y * 64 + rowPair * 32;

    const int hoff = hiHalf ? 8 : 0;
    const _Float16* aRow0 = A + (size_t)(row0 + (int)m) * K + hoff;
    const _Float16* aRow1 = aRow0 + (size_t)16 * K;
    const _Float16* wRow  = W + (size_t)(col0 + (int)m) * K + hoff;

    v8f acc[2][4];
#pragma unroll
    for (int rp = 0; rp < 2; ++rp)
#pragma unroll
        for (int c = 0; c < 4; ++c) acc[rp][c] = (v8f){};

    for (int k = 0; k < K; k += 32) {
        __builtin_prefetch(aRow0 + k + 128, 0, 1);
        __builtin_prefetch(aRow1 + k + 128, 0, 1);
        // load all six fragments first so the scheduler can overlap the
        // next iteration's loads with this iteration's WMMAs
        v16h a0 = frag_load(aRow0 + k);
        v16h a1 = frag_load(aRow1 + k);
        v16h b0 = frag_load(wRow + 0 * 16 * K + k);
        v16h b1 = frag_load(wRow + 1 * 16 * K + k);
        v16h b2 = frag_load(wRow + 2 * 16 * K + k);
        v16h b3 = frag_load(wRow + 3 * 16 * K + k);
        acc[0][0] = wmma_f16(a0, b0, acc[0][0]);
        acc[1][0] = wmma_f16(a1, b0, acc[1][0]);
        acc[0][1] = wmma_f16(a0, b1, acc[0][1]);
        acc[1][1] = wmma_f16(a1, b1, acc[1][1]);
        acc[0][2] = wmma_f16(a0, b2, acc[0][2]);
        acc[1][2] = wmma_f16(a1, b2, acc[1][2]);
        acc[0][3] = wmma_f16(a0, b3, acc[0][3]);
        acc[1][3] = wmma_f16(a1, b3, acc[1][3]);
    }

    // epilogue: C layout per tile = lanes 0-15 (M=r,N=lane), 16-31 (M=r+8)
#pragma unroll
    for (int rp = 0; rp < 2; ++rp) {
        const int rbase = row0 + rp * 16 + (hiHalf ? 8 : 0);
#pragma unroll
        for (int c = 0; c < 4; ++c) {
            const int col  = col0 + c * 16 + (int)m;
            const float bv = (flags & 1) ? bias[col] : 0.0f;
#pragma unroll
            for (int r = 0; r < 8; ++r) {
                const int row = rbase + r;
                float v = acc[rp][c][r] + bv;
                if (flags & 2) v = 0.5f * v * (1.0f + erff(v * 0.70710678118654752f));
                const size_t o = (size_t)row * NOUT + col;
                if (flags & 4) v += resid[o];
                if (out32) out32[o] = v;
                if (out16) out16[o] = (_Float16)v;
            }
        }
    }
}

template <int K, int NOUT>
static void launch_gemm(const _Float16* A, const _Float16* W, const float* bias,
                        const float* resid, float* o32, _Float16* o16,
                        int flags, hipStream_t stream) {
    dim3 grid((NOUT + 255) / 256, NL / 64);
    gemm16_wmma<K, NOUT><<<grid, 256, 0, stream>>>(A, W, bias, resid, o32, o16, flags);
}

// ---------------------------------------------------------------------------
// Front-end: Fourier embedding MLP + cat embedding + LN/ReLU + out proj.
// One 256-thread block per row (n*L + l).  All small dots done on VALU.
// ---------------------------------------------------------------------------
__global__ void embed_kernel(const float* __restrict__ pos,
                             const int* __restrict__ attr,
                             const float* __restrict__ cat_emb,
                             const float* __restrict__ int_emb,
                             const float* __restrict__ freqs,
                             const float* __restrict__ f_w1,
                             const float* __restrict__ f_b1,
                             const float* __restrict__ f_g1,
                             const float* __restrict__ f_bt1,
                             const float* __restrict__ f_w2,
                             const float* __restrict__ f_b2,
                             const float* __restrict__ out_g,
                             const float* __restrict__ out_bt,
                             const float* __restrict__ out_w,
                             const float* __restrict__ out_b,
                             float* __restrict__ X) {
    __shared__ float emb[2][132];
    __shared__ float hl[2][DD];
    __shared__ float rbuf[DD];
    __shared__ float red[256];

    const int row = blockIdx.x;         // n*L + l
    const int n   = row / LL;
    const int l   = row % LL;
    const int tid = threadIdx.x;

    const float* p = pos + (size_t)row * 2;
    float feat[2];
    if (l == 0) { feat[0] = 0.0f; feat[1] = 0.0f; }
    else        { feat[0] = p[0] - p[-2]; feat[1] = p[1] - p[-1]; }

    const float twopi = 6.283185307179586477f;
    for (int i = 0; i < 2; ++i) {
        const float ft = feat[i];
        if (tid < 64)        emb[i][tid] = cosf(ft * freqs[i * 64 + tid] * twopi);
        else if (tid < 128)  emb[i][tid] = sinf(ft * freqs[i * 64 + tid - 64] * twopi);
        else if (tid == 128) emb[i][128] = ft;
    }
    __syncthreads();

    // h[i][d] = emb[i] . f_w1[i,d,:] + f_b1 ; LN over d ; relu
    for (int i = 0; i < 2; ++i) {
        const float* w = f_w1 + ((size_t)(i * DD + tid)) * 129;
        float h = f_b1[i * DD + tid];
        for (int j = 0; j < 129; ++j) h += emb[i][j] * w[j];
        const float mean = block_sum256(h, red) * (1.0f / 256.0f);
        const float d0   = h - mean;
        float var = block_sum256(d0 * d0, red) * (1.0f / 256.0f);
        var = fmaxf(var, 0.0f);
        float y = d0 * rsqrtf(var + 1e-5f) * f_g1[i * DD + tid] + f_bt1[i * DD + tid];
        hl[i][tid] = fmaxf(y, 0.0f);
    }
    __syncthreads();

    // x[e] = sum_i ( hl[i] . f_w2[i,e,:] + f_b2[i,e] ) + cat[e]
    const int a0 = attr[n * 3 + 0];
    const int a2 = attr[n * 3 + 2];
    float x = cat_emb[(size_t)a0 * DD + tid] + int_emb[(size_t)a2 * DD + tid];
    for (int i = 0; i < 2; ++i) {
        const float* w = f_w2 + ((size_t)(i * DD + tid)) * DD;
        float acc = f_b2[i * DD + tid];
        for (int d = 0; d < DD; ++d) acc += hl[i][d] * w[d];
        x += acc;
    }
    // LN + relu over e
    {
        const float mean = block_sum256(x, red) * (1.0f / 256.0f);
        const float d0   = x - mean;
        float var = block_sum256(d0 * d0, red) * (1.0f / 256.0f);
        var = fmaxf(var, 0.0f);
        float y = d0 * rsqrtf(var + 1e-5f) * out_g[tid] + out_bt[tid];
        rbuf[tid] = fmaxf(y, 0.0f);
    }
    __syncthreads();
    // x2[e] = rbuf . out_w[e,:] + out_b[e]
    {
        const float* w = out_w + (size_t)tid * DD;
        float acc = out_b[tid];
        for (int d = 0; d < DD; ++d) acc += rbuf[d] * w[d];
        X[(size_t)row * DD + tid] = acc;
    }
}

// ---------------------------------------------------------------------------
// LayerNorm over D=256 per row; optional fp32 addend, relu, fp32/f16 outputs.
// ---------------------------------------------------------------------------
__global__ void ln_kernel(const float* __restrict__ in,
                          const float* __restrict__ add,
                          const float* __restrict__ g,
                          const float* __restrict__ b,
                          float* __restrict__ out32,
                          _Float16* __restrict__ out16,
                          int relu) {
    __shared__ float red[256];
    const size_t base = (size_t)blockIdx.x * DD;
    const int tid = threadIdx.x;
    float v = in[base + tid];
    if (add) v += add[base + tid];
    const float mean = block_sum256(v, red) * (1.0f / 256.0f);
    const float d0   = v - mean;
    float var = block_sum256(d0 * d0, red) * (1.0f / 256.0f);
    var = fmaxf(var, 0.0f);
    float y = d0 * rsqrtf(var + 1e-5f) * g[tid] + b[tid];
    if (relu) y = fmaxf(y, 0.0f);
    if (out32) out32[base + tid] = y;
    if (out16) out16[base + tid] = (_Float16)y;
}

// ---------------------------------------------------------------------------
// Attention: one wave per (n, h).  q,k,v (20 x 32) staged in LDS (stride 33),
// softmax over keys with padding bias, output o (f16).
// ---------------------------------------------------------------------------
__global__ void attn_kernel(const _Float16* __restrict__ qkv,   // NL x 768
                            const unsigned char* __restrict__ pt,
                            _Float16* __restrict__ o16) {       // NL x 256
    __shared__ float q_s[LL * 33];
    __shared__ float k_s[LL * 33];
    __shared__ float v_s[LL * 33];
    __shared__ float p_s[32];

    const int nh = blockIdx.x;
    const int n  = nh >> 3;
    const int h  = nh & 7;
    const int lane = threadIdx.x;

    const size_t rb = (size_t)n * LL * 768;
    for (int l = 0; l < LL; ++l) {
        const size_t base = rb + (size_t)l * 768 + (size_t)h * DHH + lane;
        q_s[l * 33 + lane] = (float)qkv[base];
        k_s[l * 33 + lane] = (float)qkv[base + 256];
        v_s[l * 33 + lane] = (float)qkv[base + 512];
    }
    __syncthreads();

    const float scale = 0.1767766952966368811f;   // 1/sqrt(32)
    for (int l = 0; l < LL; ++l) {
        float s = -3.0e38f;
        if (lane < LL) {
            float acc = 0.0f;
            for (int d = 0; d < DHH; ++d)
                acc += q_s[l * 33 + d] * k_s[lane * 33 + d];
            s = acc * scale + (pt[(size_t)n * LL + lane] ? NEGBIAS : 0.0f);
        }
        float mx = s;
        for (int off = 16; off; off >>= 1) mx = fmaxf(mx, __shfl_xor(mx, off, 32));
        float e = (lane < LL) ? __expf(s - mx) : 0.0f;
        float sum = e;
        for (int off = 16; off; off >>= 1) sum += __shfl_xor(sum, off, 32);
        p_s[lane] = e / sum;
        __syncthreads();
        float acc = 0.0f;
        for (int j = 0; j < LL; ++j) acc += p_s[j] * v_s[j * 33 + lane];
        o16[((size_t)n * LL + l) * DD + (size_t)h * DHH + lane] = (_Float16)acc;
        __syncthreads();
    }
}

// ---------------------------------------------------------------------------
// max over L per (n,d)
// ---------------------------------------------------------------------------
__global__ void rowmax_kernel(const float* __restrict__ t, float* __restrict__ mx) {
    const int n = blockIdx.x, d = threadIdx.x;
    float m = -3.0e38f;
    for (int l = 0; l < LL; ++l)
        m = fmaxf(m, t[((size_t)n * LL + l) * DD + d]);
    mx[(size_t)n * DD + d] = m;
}

// build cat2 = [t, broadcast(mx)] as f16 (NL x 512)
__global__ void catbuild_kernel(const float* __restrict__ t,
                                const float* __restrict__ mx,
                                _Float16* __restrict__ cat16) {
    const size_t row = blockIdx.x;
    const int tid = threadIdx.x;
    const int n = (int)(row / LL);
    cat16[row * 512 + tid]       = (_Float16)t[row * DD + tid];
    cat16[row * 512 + 256 + tid] = (_Float16)mx[(size_t)n * DD + tid];
}

// final pooled output with key-padding zeroing
__global__ void pool_kernel(const float* __restrict__ x,
                            const unsigned char* __restrict__ keymask,
                            float* __restrict__ out) {
    const int n = blockIdx.x, d = threadIdx.x;
    float m = -3.0e38f;
    for (int l = 0; l < LL; ++l)
        m = fmaxf(m, x[((size_t)n * LL + l) * DD + d]);
    out[(size_t)n * DD + d] = keymask[n] ? 0.0f : m;
}

__global__ void f32_to_f16_kernel(const float* __restrict__ s,
                                  _Float16* __restrict__ d, int count) {
    const int i = blockIdx.x * 256 + threadIdx.x;
    if (i < count) d[i] = (_Float16)s[i];
}

// ---------------------------------------------------------------------------
// launch
// ---------------------------------------------------------------------------
extern "C" void kernel_launch(void* const* d_in, const int* in_sizes, int n_in,
                              void* d_out, int out_size, void* d_ws, size_t ws_size,
                              hipStream_t stream) {
    (void)in_sizes; (void)n_in; (void)out_size; (void)ws_size;

    // inputs in setup_inputs() dict order
    const float* lane_positions = (const float*)d_in[0];
    const int*   lane_attr      = (const int*)d_in[1];
    const unsigned char* pt_mask  = (const unsigned char*)d_in[2];
    const unsigned char* key_mask = (const unsigned char*)d_in[3];
    const float* cat_emb = (const float*)d_in[4];
    const float* int_emb = (const float*)d_in[5];
    const float* freqs   = (const float*)d_in[6];
    const float* f_w1 = (const float*)d_in[7];
    const float* f_b1 = (const float*)d_in[8];
    const float* f_g1 = (const float*)d_in[9];
    const float* f_bt1 = (const float*)d_in[10];
    const float* f_w2 = (const float*)d_in[11];
    const float* f_b2 = (const float*)d_in[12];
    const float* out_g = (const float*)d_in[13];
    const float* out_bt = (const float*)d_in[14];
    const float* out_w = (const float*)d_in[15];
    const float* out_b = (const float*)d_in[16];
    const float* n1_g = (const float*)d_in[17];
    const float* n1_b = (const float*)d_in[18];
    const float* qkv_w = (const float*)d_in[19];
    const float* qkv_b = (const float*)d_in[20];
    const float* proj_w = (const float*)d_in[21];
    const float* proj_b = (const float*)d_in[22];
    const float* n2_g = (const float*)d_in[23];
    const float* n2_b = (const float*)d_in[24];
    const float* fc1_w = (const float*)d_in[25];
    const float* fc1_b = (const float*)d_in[26];
    const float* fc2_w = (const float*)d_in[27];
    const float* fc2_b = (const float*)d_in[28];
    const float* tn_g = (const float*)d_in[29];
    const float* tn_b = (const float*)d_in[30];
    const float* ft1_w = (const float*)d_in[31];
    const float* ft1_b = (const float*)d_in[32];
    const float* ft1_g = (const float*)d_in[33];
    const float* ft1_bt = (const float*)d_in[34];
    const float* ft2_w = (const float*)d_in[35];
    const float* ft2_b = (const float*)d_in[36];
    const float* ft2_g = (const float*)d_in[37];
    const float* ft2_bt = (const float*)d_in[38];

    float* out = (float*)d_out;
    char* ws = (char*)d_ws;

    // ------ workspace layout ------
    const size_t nl = (size_t)NL;
    size_t off = 0;
    float* X  = (float*)(ws + off); off += nl * DD * 4;     // activations
    float* T  = (float*)(ws + off); off += nl * DD * 4;     // residual stream
    float* U  = (float*)(ws + off); off += nl * DD * 4;     // fuse temp
    float* MX = (float*)(ws + off); off += (size_t)NN * DD * 4;
    _Float16* A16 = (_Float16*)(ws + off); off += nl * DD * 2;          // narrow f16
    _Float16* B16 = (_Float16*)(ws + off); off += nl * (size_t)FF_ * 2; // wide f16
    _Float16* Wqkv  = (_Float16*)(ws + off); off += (size_t)3 * 768 * 256 * 2;
    _Float16* Wproj = (_Float16*)(ws + off); off += (size_t)3 * 256 * 256 * 2;
    _Float16* Wfc1  = (_Float16*)(ws + off); off += (size_t)3 * 1024 * 256 * 2;
    _Float16* Wfc2  = (_Float16*)(ws + off); off += (size_t)3 * 256 * 1024 * 2;
    _Float16* Wft1  = (_Float16*)(ws + off); off += (size_t)3 * 256 * 512 * 2;
    _Float16* Wft2  = (_Float16*)(ws + off); off += (size_t)3 * 256 * 256 * 2;

    // ------ weight conversion (f32 -> f16), once per launch ------
    auto conv = [&](const float* s, _Float16* d, int count) {
        f32_to_f16_kernel<<<(count + 255) / 256, 256, 0, stream>>>(s, d, count);
    };
    conv(qkv_w,  Wqkv,  3 * 768 * 256);
    conv(proj_w, Wproj, 3 * 256 * 256);
    conv(fc1_w,  Wfc1,  3 * 1024 * 256);
    conv(fc2_w,  Wfc2,  3 * 256 * 1024);
    conv(ft1_w,  Wft1,  3 * 256 * 512);
    conv(ft2_w,  Wft2,  3 * 256 * 256);

    // ------ front-end ------
    embed_kernel<<<NL, 256, 0, stream>>>(lane_positions, lane_attr, cat_emb,
                                         int_emb, freqs, f_w1, f_b1, f_g1, f_bt1,
                                         f_w2, f_b2, out_g, out_bt, out_w, out_b, X);

    // ------ transformer layers ------
    for (int i = 0; i < 3; ++i) {
        // h1 = LN(x) -> f16
        ln_kernel<<<NL, 256, 0, stream>>>(X, nullptr, n1_g + i * DD, n1_b + i * DD,
                                          nullptr, A16, 0);
        // qkv = h1 @ qkv_w^T + b   -> f16 (B16)
        launch_gemm<256, 768>(A16, Wqkv + (size_t)i * 768 * 256, qkv_b + i * 768,
                              nullptr, nullptr, B16, 1, stream);
        // attention -> o16 (A16)
        attn_kernel<<<NN * HH, 32, 0, stream>>>(B16, pt_mask, A16);
        // t = x + o @ proj_w^T + b   -> fp32 (T)
        launch_gemm<256, 256>(A16, Wproj + (size_t)i * 256 * 256, proj_b + i * DD,
                              X, T, nullptr, 1 | 4, stream);
        // h2 = LN(t) -> f16
        ln_kernel<<<NL, 256, 0, stream>>>(T, nullptr, n2_g + i * DD, n2_b + i * DD,
                                          nullptr, A16, 0);
        // ff = gelu(h2 @ fc1^T + b) -> f16 (B16)
        launch_gemm<256, 1024>(A16, Wfc1 + (size_t)i * 1024 * 256, fc1_b + i * FF_,
                               nullptr, nullptr, B16, 1 | 2, stream);
        // t = t + ff @ fc2^T + b   (in-place residual)
        launch_gemm<1024, 256>(B16, Wfc2 + (size_t)i * 256 * 1024, fc2_b + i * DD,
                               T, T, nullptr, 1 | 4, stream);
        // mx = max_l t ; cat2 = [t, mx] -> f16 (B16, width 512)
        rowmax_kernel<<<NN, 256, 0, stream>>>(T, MX);
        catbuild_kernel<<<NL, 256, 0, stream>>>(T, MX, B16);
        // f1 = cat2 @ ft1^T + b -> fp32 (T, T is dead now)
        launch_gemm<512, 256>(B16, Wft1 + (size_t)i * 256 * 512, ft1_b + i * DD,
                              nullptr, T, nullptr, 1, stream);
        // relu(LN(f1)) -> f16 (A16)
        ln_kernel<<<NL, 256, 0, stream>>>(T, nullptr, ft1_g + i * DD, ft1_bt + i * DD,
                                          nullptr, A16, 1);
        // f2 = . @ ft2^T + b -> fp32 (U)
        launch_gemm<256, 256>(A16, Wft2 + (size_t)i * 256 * 256, ft2_b + i * DD,
                              nullptr, U, nullptr, 1, stream);
        // f = relu(LN(f2)) (in-place U)
        ln_kernel<<<NL, 256, 0, stream>>>(U, nullptr, ft2_g + i * DD, ft2_bt + i * DD,
                                          U, nullptr, 1);
        // x = LN(x0 + f) (in-place X)
        ln_kernel<<<NL, 256, 0, stream>>>(X, U, tn_g + i * DD, tn_b + i * DD,
                                          X, nullptr, 0);
    }

    // ------ pooled output ------
    pool_kernel<<<NN, 256, 0, stream>>>(X, key_mask, out);
}
